// CrfDecoderABC_88244398063964
// MI455X (gfx1250) — matrix operand order repaired and checked by reading, not compiled, not run on hardware
//
#include <hip/hip_runtime.h>
#include <hip/hip_fp16.h>

// ---------------------------------------------------------------------------
// Linear-chain CRF log-partition on MI455X (gfx1250).
//
// Phase 1 (crf_chunk_kernel): for each (batch b, time-chunk c) compute the
//   64x64 chunk operator  P_c = prod_{t in chunk, t < len_b} G_t,
//   G_t[i][j] = exp(transitions[i][j]) * exp(emissions[b,t,j]),
//   in rescaled linear space: each step renormalizes every row of the running
//   product by its max (log of the scale accumulated per row in s[]), keeping
//   f16 operands in (0,1].  The 64x64 x 64x64 step matmul runs on
//   v_wmma_f32_16x16x32_f16 (4 waves, each owning 16 rows: 4 col-tiles x 2
//   K-chunks = 8 WMMAs/step, f32 accumulation).  The diag(exp(emit)) factor is
//   applied to the WMMA output columns (lane == column in C layout), so the
//   expT B-fragments are built once and stay resident in VGPRs.
//   Emissions for the chunk are a contiguous [steps x 64] f32 slab -> TDM
//   tensor_load_to_lds double-buffer prefetch, waited with s_wait_tensorcnt.
//
// Phase 2 (crf_combine_kernel): per batch, short sequential scan over the C
//   chunk operators in log space:
//   alpha_j <- m + log( sum_i exp(alpha_i + s_c[i] - m) * P~_c[i][j] ),
//   then logsumexp with end_transitions.
// ---------------------------------------------------------------------------

typedef _Float16 v16h __attribute__((ext_vector_type(16)));
typedef float    v8f  __attribute__((ext_vector_type(8)));
typedef unsigned int v4u __attribute__((ext_vector_type(4)));
typedef int      v8i  __attribute__((ext_vector_type(8)));
typedef int      v4i  __attribute__((ext_vector_type(4)));

#define B_BATCH 256
#define T_LEN   2048
#define N_TAG   64
#define PREF    16            // timesteps per TDM prefetch buffer
#define LDSPAD  8             // f16 pad per row -> 144B row stride (16B aligned)

// Issue one TDM load of a contiguous f32 slab (PREF*64 elements) into LDS.
// D# per CDNA5 ISA: 1 descriptor, data_size=4B,
// tile_dim0 = PREF*64, tensor_dim0 = remaining valid elements (OOB -> zero).
// This toolchain exposes the 6-arg builtin: (g0, g1, g2, g3, g2b, cpol).
__device__ __forceinline__ void tdm_load_slab(const float* gsrc,
                                              unsigned lds_off,
                                              unsigned elems_valid) {
  unsigned long long ga = (unsigned long long)(size_t)gsrc;
  v4u g0;
  g0[0] = 1u;                                           // count=1, user mode
  g0[1] = lds_off;                                      // LDS byte address
  g0[2] = (unsigned)(ga & 0xffffffffu);                 // global_addr[31:0]
  g0[3] = (unsigned)((ga >> 32) & 0x01ffffffu) | (2u << 30); // addr[56:32] | type=2
  v8i g1;
  g1[0] = (int)(2u << 16);                              // data_size = 4 bytes
  g1[1] = (int)((elems_valid & 0xffffu) << 16);         // tensor_dim0[15:0]
  g1[2] = (int)(((elems_valid >> 16) & 0xffffu) | (1u << 16)); // dim0 hi | tensor_dim1=1
  g1[3] = (int)((unsigned)(PREF * N_TAG) << 16);        // tile_dim0
  g1[4] = 0;                                            // tile_dim1/2 unused (1D)
  g1[5] = (int)(PREF * N_TAG);                          // tensor_dim0_stride lo
  g1[6] = 0;
  g1[7] = 0;
  v4i z4; z4[0] = 0; z4[1] = 0; z4[2] = 0; z4[3] = 0;   // groups 2/3 unused (<=2D)
  v8i z8; z8[0] = 0; z8[1] = 0; z8[2] = 0; z8[3] = 0;
  z8[4] = 0; z8[5] = 0; z8[6] = 0; z8[7] = 0;
  __builtin_amdgcn_tensor_load_to_lds(g0, g1, z4, z4, z8, 0);
}

__launch_bounds__(128, 1)
__global__ void crf_chunk_kernel(const float* __restrict__ emissions,
                                 const float* __restrict__ transitions,
                                 const int*   __restrict__ lengths,
                                 _Float16*    __restrict__ Pws,
                                 float*       __restrict__ Sws,
                                 int C, int L) {
  __shared__ float    lds_emis[2][PREF * N_TAG];              // 8 KB TDM buffers
  __shared__ _Float16 lds_P[2][N_TAG][N_TAG + LDSPAD];        // 18 KB ping-pong

  const int c    = blockIdx.x;
  const int b    = blockIdx.y;
  const int tid  = threadIdx.x;
  const int w    = tid >> 5;        // wave 0..3: owns output rows [16w,16w+16)
  const int lane = tid & 31;
  const int li   = lane & 15;       // lane within half
  const int hi   = lane >> 4;       // lane half

  const int t0 = 1 + c * L;
  int t1 = t0 + L; if (t1 > T_LEN) t1 = T_LEN;
  const int span = t1 - t0;
  int vsteps = lengths[b] - t0;     // steps with t < len (rest are identity)
  if (vsteps < 0) vsteps = 0;
  if (vsteps > span) vsteps = span;

  // ---- constant B fragments: expT, f16, ISA B-layout (lane=col, K striped) --
  v16h bf[4][2];
#pragma unroll
  for (int ct = 0; ct < 4; ++ct) {
    const int col = li + 16 * ct;
#pragma unroll
    for (int kc = 0; kc < 2; ++kc)
#pragma unroll
      for (int e = 0; e < 16; ++e) {
        const int K = 32 * kc + 8 * hi + (e & 7) + 16 * (e >> 3);
        bf[ct][kc][e] = (_Float16)__expf(transitions[K * N_TAG + col]);
      }
  }

  // ---- A fragments: running product P, init = identity rows [16w..16w+16) --
  const int row_g = 16 * w + li;    // global row this lane holds in A layout
  v16h af[2];
#pragma unroll
  for (int kc = 0; kc < 2; ++kc)
#pragma unroll
    for (int e = 0; e < 16; ++e) {
      const int K = 32 * kc + 8 * hi + (e & 7) + 16 * (e >> 3);
      af[kc][e] = (_Float16)((K == row_g) ? 1.0f : 0.0f);
    }
  float s_reg[8];                   // per-row log scales (rows r+8*hi+16w)
#pragma unroll
  for (int r = 0; r < 8; ++r) s_reg[r] = 0.0f;

  // ---- prime TDM prefetch of first emission slab -----------------------
  if (vsteps > 0 && w == 0) {
    tdm_load_slab(emissions + ((size_t)b * T_LEN + t0) * N_TAG,
                  (unsigned)(size_t)&lds_emis[0][0],
                  (unsigned)((T_LEN - t0) * N_TAG));
  }

  for (int step = 0; step < vsteps; ++step) {
    const int ph  = step & (PREF - 1);
    const int buf = (step / PREF) & 1;
    if (ph == 0) {
      __syncthreads();                              // old buffer fully consumed
      if (w == 0) {
        __builtin_amdgcn_s_wait_tensorcnt(0);       // current buffer landed
        const int tn = t0 + step + PREF;
        if (tn < t0 + vsteps) {
          tdm_load_slab(emissions + ((size_t)b * T_LEN + tn) * N_TAG,
                        (unsigned)(size_t)&lds_emis[buf ^ 1][0],
                        (unsigned)((T_LEN - tn) * N_TAG));
        }
      }
      __syncthreads();                              // publish current buffer
    }

    // diag(exp(emit_t)) column factors (lane == column in C layout)
    float ef[4];
#pragma unroll
    for (int ct = 0; ct < 4; ++ct)
      ef[ct] = __expf(lds_emis[buf][ph * N_TAG + li + 16 * ct]);

    // P_new rows = P rows x expT  (8 chained WMMAs)
    v8f acc[4];
#pragma unroll
    for (int ct = 0; ct < 4; ++ct) {
      v8f cz = {0.f, 0.f, 0.f, 0.f, 0.f, 0.f, 0.f, 0.f};
      cz = __builtin_amdgcn_wmma_f32_16x16x32_f16(false, af[0], false, bf[ct][0],
                                                  (short)0, cz, false, false);
      cz = __builtin_amdgcn_wmma_f32_16x16x32_f16(false, af[1], false, bf[ct][1],
                                                  (short)0, cz, false, false);
      acc[ct] = cz;
    }

    // scale columns by exp(emit), per-row max across 4 tiles then 16 lanes
    float rm[8];
#pragma unroll
    for (int r = 0; r < 8; ++r) {
      float m = 0.0f;
#pragma unroll
      for (int ct = 0; ct < 4; ++ct) {
        acc[ct][r] *= ef[ct];
        m = fmaxf(m, acc[ct][r]);
      }
      m = fmaxf(m, __shfl_xor(m, 1));
      m = fmaxf(m, __shfl_xor(m, 2));
      m = fmaxf(m, __shfl_xor(m, 4));
      m = fmaxf(m, __shfl_xor(m, 8));
      rm[r] = m;
    }

    // renormalize rows, fold scale into s, stage f16 P_new in LDS (ping-pong)
    const int pp = step & 1;
#pragma unroll
    for (int r = 0; r < 8; ++r) {
      const float rinv = __builtin_amdgcn_rcpf(rm[r]);
      s_reg[r] += __logf(rm[r]);
      const int grow = 16 * w + 8 * hi + r;
#pragma unroll
      for (int ct = 0; ct < 4; ++ct)
        lds_P[pp][grow][li + 16 * ct] = (_Float16)(acc[ct][r] * rinv);
    }
    __syncthreads();

    // reload A fragments (lane=row, K runs [k0,k0+8) and [k0+16,k0+24))
#pragma unroll
    for (int kc = 0; kc < 2; ++kc) {
      const int k0 = 32 * kc + 8 * hi;
      const _Float16* src = &lds_P[pp][row_g][0];
#pragma unroll
      for (int e = 0; e < 8; ++e) {
        af[kc][e]     = src[k0 + e];
        af[kc][8 + e] = src[k0 + 16 + e];
      }
    }
  }

  // ---- emit chunk operator (f16 matrix + f32 row log-scales) ----------
  _Float16* Pout = Pws + ((size_t)b * C + c) * (N_TAG * N_TAG);
  float*    Sout = Sws + ((size_t)b * C + c) * N_TAG;
  if (vsteps == 0) {
    for (int idx = tid; idx < N_TAG * N_TAG; idx += 128)
      Pout[idx] = (_Float16)(((idx >> 6) == (idx & 63)) ? 1.0f : 0.0f);
    if (tid < N_TAG) Sout[tid] = 0.0f;
  } else {
    const int pp = (vsteps - 1) & 1;
    for (int idx = tid; idx < N_TAG * N_TAG; idx += 128)
      Pout[idx] = lds_P[pp][idx >> 6][idx & 63];
    if (li == 0) {
#pragma unroll
      for (int r = 0; r < 8; ++r)
        Sout[16 * w + 8 * hi + r] = s_reg[r];
    }
  }
}

__global__ void crf_combine_kernel(const float* __restrict__ emissions,
                                   const float* __restrict__ startp,
                                   const float* __restrict__ endp,
                                   const _Float16* __restrict__ Pws,
                                   const float* __restrict__ Sws,
                                   float* __restrict__ out, int C) {
  __shared__ float sa[N_TAG];
  __shared__ float red[N_TAG];
  const int b = blockIdx.x;
  const int j = threadIdx.x;                // one tag column per thread

  float alpha = startp[j] + emissions[(size_t)b * T_LEN * N_TAG + j];

  for (int c = 0; c < C; ++c) {
    const _Float16* P = Pws + ((size_t)b * C + c) * (N_TAG * N_TAG);
    const float*    S = Sws + ((size_t)b * C + c) * N_TAG;
    const float beta = alpha + S[j];
    red[j] = beta; __syncthreads();
    for (int off = 32; off >= 1; off >>= 1) {
      if (j < off) red[j] = fmaxf(red[j], red[j + off]);
      __syncthreads();
    }
    const float m = red[0]; __syncthreads();
    sa[j] = __expf(beta - m); __syncthreads();
    float v = 0.0f;
#pragma unroll 8
    for (int i = 0; i < N_TAG; ++i)
      v += sa[i] * (float)P[i * N_TAG + j];
    alpha = m + __logf(v);
    __syncthreads();
  }

  const float z = alpha + endp[j];
  red[j] = z; __syncthreads();
  for (int off = 32; off >= 1; off >>= 1) {
    if (j < off) red[j] = fmaxf(red[j], red[j + off]);
    __syncthreads();
  }
  const float m = red[0]; __syncthreads();
  red[j] = __expf(z - m); __syncthreads();
  for (int off = 32; off >= 1; off >>= 1) {
    if (j < off) red[j] += red[j + off];
    __syncthreads();
  }
  if (j == 0) out[b] = m + __logf(red[0]);
}

extern "C" void kernel_launch(void* const* d_in, const int* in_sizes, int n_in,
                              void* d_out, int out_size, void* d_ws, size_t ws_size,
                              hipStream_t stream) {
  (void)in_sizes; (void)n_in; (void)out_size;
  const float* emissions   = (const float*)d_in[0];
  const float* transitions = (const float*)d_in[1];
  const float* startp      = (const float*)d_in[2];
  const float* endp        = (const float*)d_in[3];
  const int*   lengths     = (const int*)d_in[4];

  // workspace per chunk-operator: 64x64 f16 matrix + 64 f32 row scales
  const size_t per_chunk = (size_t)B_BATCH * (N_TAG * N_TAG * 2 + N_TAG * 4);
  int C = 16;
  while (C > 1 && per_chunk * (size_t)C > ws_size) C >>= 1;
  const int L = (T_LEN - 1 + C - 1) / C;

  _Float16* Pws = (_Float16*)d_ws;
  float*    Sws = (float*)((char*)d_ws + (size_t)B_BATCH * C * N_TAG * N_TAG * 2);

  crf_chunk_kernel<<<dim3(C, B_BATCH), 128, 0, stream>>>(
      emissions, transitions, lengths, Pws, Sws, C, L);
  crf_combine_kernel<<<B_BATCH, N_TAG, 0, stream>>>(
      emissions, startp, endp, Pws, Sws, (float*)d_out, C);
}
